// Decoder_gru_8589935085
// MI455X (gfx1250) — matrix operand order repaired and checked by reading, not compiled, not run on hardware
//
#include <hip/hip_runtime.h>
#include <cstddef>

typedef __attribute__((ext_vector_type(16))) _Float16 v16h;
typedef __attribute__((ext_vector_type(8)))  _Float16 v8h;
typedef __attribute__((ext_vector_type(8)))  float    v8f;

#define NN   84
#define PN   3486
#define PPAD 3488
#define HSZ  2048
#define H2   1024
#define G3H  6144

__device__ __forceinline__ float sigm_(float v) {
    return 1.0f / (1.0f + __expf(-v));
}
__device__ __forceinline__ float tanh_(float v) {
    float t = __expf(-2.0f * fabsf(v));
    float r = (1.0f - t) / (1.0f + t);
    return v < 0.0f ? -r : r;
}

__device__ __forceinline__ v8f wmma16(v16h a, v16h b, v8f c) {
    return __builtin_amdgcn_wmma_f32_16x16x32_f16(
        /*neg_a=*/false, a, /*neg_b=*/false, b,
        /*c_mod=*/(short)0, c, /*reuse_a=*/false, /*reuse_b=*/false);
}

__device__ __forceinline__ v16h cat8(v8h lo, v8h hi) {
    return __builtin_shufflevector(lo, hi, 0,1,2,3,4,5,6,7,8,9,10,11,12,13,14,15);
}

// ---------------------------------------------------------------- prep kernels

__global__ void k_pairs(int2* __restrict__ pairs) {
    int p = blockIdx.x * blockDim.x + threadIdx.x;
    if (p >= PN) return;
    int i = 0, c = 0;
    while (p >= c + (NN - 1 - i)) { c += NN - 1 - i; ++i; }
    int j = i + 1 + (p - c);
    pairs[p] = make_int2(i, j);
}

__global__ void k_cvt(const float* __restrict__ src, _Float16* __restrict__ dst, int n) {
    int i = blockIdx.x * blockDim.x + threadIdx.x;
    if (i < n) dst[i] = (_Float16)src[i];
}

// gin[n, j] = x[n,:64] . Wih[j,:64] + bih[j]
__global__ void k_gin(const float* __restrict__ x, const float* __restrict__ Wih,
                      const float* __restrict__ bih, float* __restrict__ gin) {
    int idx = blockIdx.x * blockDim.x + threadIdx.x;
    if (idx >= NN * G3H) return;
    int n = idx / G3H, j = idx % G3H;
    const float* xr = x + n * 64;
    const float* wr = Wih + (size_t)j * 64;
    float s = bih[j];
#pragma unroll
    for (int k = 0; k < 64; ++k) s += xr[k] * wr[k];
    gin[idx] = s;
}

// ---------------------------------------------------------------- fused GRU GEMM
// One wave per block. Output tile: 32 rows (pairs) x 16 cols (hidden j).
// acc_g[p,j] = sum_k h[p,k] * Whh[g*2048 + j, k], g in {r,z,n}; epilogue = GRU cell.
__global__ __launch_bounds__(32)
void k_gru(const _Float16* __restrict__ A,     // [PPAD, HSZ] f16 state
           const float*    __restrict__ Hold,  // [PN, HSZ] f32 previous h
           const _Float16* __restrict__ Whh,   // [G3H, HSZ] f16
           const float*    __restrict__ bhh,   // [G3H]
           const float*    __restrict__ gin,   // [NN, G3H]
           const int2*     __restrict__ pairs, // [PN]
           int sel,                            // 0 -> iu, 1 -> ju
           float*          __restrict__ Hs,    // out f32 state
           _Float16*       __restrict__ Hh)    // out f16 state
{
    const int lane  = threadIdx.x;
    const int pbase = blockIdx.x * 32;
    const int jbase = blockIdx.y * 16;
    const int mrow  = lane & 15;
    const int khalf = lane >> 4;
    const int j     = jbase + mrow;   // B column / epilogue column for this lane

    const _Float16* a0 = A + (size_t)(pbase + mrow)      * HSZ;
    const _Float16* a1 = A + (size_t)(pbase + 16 + mrow) * HSZ;
    const _Float16* br = Whh + (size_t)(0    + j) * HSZ + khalf * 16;
    const _Float16* bz = Whh + (size_t)(2048 + j) * HSZ + khalf * 16;
    const _Float16* bn = Whh + (size_t)(4096 + j) * HSZ + khalf * 16;

    v8f ar[2] = {}, az[2] = {}, an[2] = {};

    for (int k = 0; k < HSZ; k += 32) {
        __builtin_prefetch(br + k + 256, 0, 1);
        __builtin_prefetch(bz + k + 256, 0, 1);
        __builtin_prefetch(bn + k + 256, 0, 1);
        v16h at0 = *(const v16h*)(a0 + k + khalf * 16);
        v16h at1 = *(const v16h*)(a1 + k + khalf * 16);
        v16h tr = *(const v16h*)(br + k);
        v16h tz = *(const v16h*)(bz + k);
        v16h tn = *(const v16h*)(bn + k);
        ar[0] = wmma16(at0, tr, ar[0]);  ar[1] = wmma16(at1, tr, ar[1]);
        az[0] = wmma16(at0, tz, az[0]);  az[1] = wmma16(at1, tz, az[1]);
        an[0] = wmma16(at0, tn, an[0]);  an[1] = wmma16(at1, tn, an[1]);
    }

    const float bhr = bhh[j], bhz = bhh[2048 + j], bhn = bhh[4096 + j];
#pragma unroll
    for (int t = 0; t < 2; ++t) {
#pragma unroll
        for (int v = 0; v < 8; ++v) {
            int p = pbase + t * 16 + v + 8 * khalf;   // D row (per C/D VGPR layout)
            if (p < PN) {
                int2 pr = pairs[p];
                int node = sel ? pr.y : pr.x;
                const float* g = gin + (size_t)node * G3H + j;
                float hr = (t ? ar[1][v] : ar[0][v]) + bhr;
                float hz = (t ? az[1][v] : az[0][v]) + bhz;
                float hn = (t ? an[1][v] : an[0][v]) + bhn;
                float r  = sigm_(g[0]    + hr);
                float z  = sigm_(g[2048] + hz);
                float n  = tanh_(g[4096] + r * hn);
                float ho = Hold[(size_t)p * HSZ + j];
                float hv = (1.0f - z) * n + z * ho;
                Hs[(size_t)p * HSZ + j] = hv;
                Hh[(size_t)p * HSZ + j] = (_Float16)hv;
            }
        }
    }
}

// ---------------------------------------------------------------- fused MLP GEMM
// 2x2 register blocking: 32 rows x 32 cols per wave. y = relu(A @ W^T + b);
// emits per-wave (sum, sumsq) partials for the whole-tensor LayerNorm.
__global__ __launch_bounds__(32)
void k_mlp(const _Float16* __restrict__ A,   // [PPAD, K] f16
           const _Float16* __restrict__ W,   // [O, K] f16
           const float*    __restrict__ bias,// [O]
           int K, int O,
           float* __restrict__ y,            // [PPAD, O] f32 (rows < PN written)
           float* __restrict__ partials)     // [2 * gridDim.x*gridDim.y]
{
    const int lane  = threadIdx.x;
    const int pbase = blockIdx.x * 32;
    const int obase = blockIdx.y * 32;
    const int mrow  = lane & 15;
    const int khalf = lane >> 4;
    const int oc0   = obase + mrow;
    const int oc1   = obase + 16 + mrow;

    const _Float16* a0 = A + (size_t)(pbase + mrow)      * K;
    const _Float16* a1 = A + (size_t)(pbase + 16 + mrow) * K;
    const _Float16* b0 = W + (size_t)oc0 * K + khalf * 16;
    const _Float16* b1 = W + (size_t)oc1 * K + khalf * 16;

    v8f acc00 = {}, acc01 = {}, acc10 = {}, acc11 = {};
    for (int k = 0; k < K; k += 32) {
        __builtin_prefetch(b0 + k + 256, 0, 1);
        __builtin_prefetch(b1 + k + 256, 0, 1);
        v16h at0 = *(const v16h*)(a0 + k + khalf * 16);
        v16h at1 = *(const v16h*)(a1 + k + khalf * 16);
        v16h bt0 = *(const v16h*)(b0 + k);
        v16h bt1 = *(const v16h*)(b1 + k);
        acc00 = wmma16(at0, bt0, acc00);
        acc10 = wmma16(at1, bt0, acc10);
        acc01 = wmma16(at0, bt1, acc01);
        acc11 = wmma16(at1, bt1, acc11);
    }

    const float bj0 = bias[oc0];
    const float bj1 = bias[oc1];
    float s = 0.0f, s2 = 0.0f;
#pragma unroll
    for (int t = 0; t < 2; ++t) {
#pragma unroll
        for (int u = 0; u < 2; ++u) {
            const int oc = u ? oc1 : oc0;
            const float bj = u ? bj1 : bj0;
#pragma unroll
            for (int v = 0; v < 8; ++v) {
                int p = pbase + t * 16 + v + 8 * khalf;
                if (p < PN) {
                    float val = (t ? (u ? acc11[v] : acc10[v])
                                   : (u ? acc01[v] : acc00[v])) + bj;
                    val = fmaxf(val, 0.0f);
                    y[(size_t)p * O + oc] = val;
                    s  += val;
                    s2 += val * val;
                }
            }
        }
    }
#pragma unroll
    for (int ofs = 16; ofs > 0; ofs >>= 1) {
        s  += __shfl_xor(s,  ofs, 32);
        s2 += __shfl_xor(s2, ofs, 32);
    }
    if (lane == 0) {
        int wg = blockIdx.y * gridDim.x + blockIdx.x;
        partials[2 * wg]     = s;
        partials[2 * wg + 1] = s2;
    }
}

// single-block deterministic reduction -> stats = {mean, inv_std}
__global__ void k_reduce(const float* __restrict__ partials, int nwg,
                         float count, float* __restrict__ stats) {
    __shared__ double sh[256], sh2[256];
    int tid = threadIdx.x;
    double s = 0.0, s2 = 0.0;
    for (int i = tid; i < nwg; i += 256) {
        s  += (double)partials[2 * i];
        s2 += (double)partials[2 * i + 1];
    }
    sh[tid] = s; sh2[tid] = s2;
    __syncthreads();
    for (int o = 128; o > 0; o >>= 1) {
        if (tid < o) { sh[tid] += sh[tid + o]; sh2[tid] += sh2[tid + o]; }
        __syncthreads();
    }
    if (tid == 0) {
        double mu  = sh[0] / (double)count;
        double var = sh2[0] / (double)count - mu * mu;
        stats[0] = (float)mu;
        stats[1] = (float)(1.0 / sqrt(var + 1e-5));
    }
}

// whole-tensor LayerNorm -> f16 operand for next GEMM
__global__ void k_ln(const float* __restrict__ y, const float* __restrict__ lnw,
                     const float* __restrict__ lnb, const float* __restrict__ stats,
                     int total, _Float16* __restrict__ out) {
    int i = blockIdx.x * blockDim.x + threadIdx.x;
    if (i >= total) return;
    float mu = stats[0], inv = stats[1];
    out[i] = (_Float16)((y[i] - mu) * inv * lnw[i] + lnb[i]);
}

// final head + symmetric scatter, diagonal zero
__global__ void k_out(const _Float16* __restrict__ h, const float* __restrict__ W4,
                      const float* __restrict__ b4, float* __restrict__ out) {
    int idx = blockIdx.x * blockDim.x + threadIdx.x;
    if (idx >= NN * NN) return;
    int a = idx / NN, b = idx % NN;
    if (a == b) { out[idx] = 0.0f; return; }
    int i = a < b ? a : b;
    int j = a < b ? b : a;
    int p = i * NN - (i * (i + 1)) / 2 + (j - i - 1);
    const _Float16* hp = h + (size_t)p * H2;
    float s = b4[0];
    for (int k = 0; k < H2; ++k) s += (float)hp[k] * W4[k];
    out[idx] = sigm_(s);
}

// ---------------------------------------------------------------- launch

extern "C" void kernel_launch(void* const* d_in, const int* in_sizes, int n_in,
                              void* d_out, int out_size, void* d_ws, size_t ws_size,
                              hipStream_t stream) {
    const float* x    = (const float*)d_in[0];
    const float* hid  = (const float*)d_in[1];
    const float* Wih  = (const float*)d_in[2];
    const float* Whh  = (const float*)d_in[3];
    const float* bih  = (const float*)d_in[4];
    const float* bhh  = (const float*)d_in[5];
    const float* W1   = (const float*)d_in[6];
    const float* b1   = (const float*)d_in[7];
    const float* ln1w = (const float*)d_in[8];
    const float* ln1b = (const float*)d_in[9];
    const float* W2   = (const float*)d_in[10];
    const float* b2   = (const float*)d_in[11];
    const float* ln2w = (const float*)d_in[12];
    const float* ln2b = (const float*)d_in[13];
    const float* W3   = (const float*)d_in[14];
    const float* b3   = (const float*)d_in[15];
    const float* ln3w = (const float*)d_in[16];
    const float* ln3b = (const float*)d_in[17];
    const float* W4   = (const float*)d_in[18];
    const float* b4   = (const float*)d_in[19];

    char* ws = (char*)d_ws;
    size_t off = 0;
    auto take = [&](size_t bytes) -> char* {
        char* p = ws + off;
        off += (bytes + 255) & ~(size_t)255;
        return p;
    };
    _Float16* WhhH = (_Float16*)take((size_t)G3H * HSZ * 2);
    _Float16* W1H  = (_Float16*)take((size_t)HSZ * HSZ * 2);
    _Float16* W2H  = (_Float16*)take((size_t)H2  * HSZ * 2);
    _Float16* W3H  = (_Float16*)take((size_t)H2  * H2  * 2);
    _Float16* hA   = (_Float16*)take((size_t)PPAD * HSZ * 2);
    _Float16* hB   = (_Float16*)take((size_t)PPAD * HSZ * 2);
    float*    hS   = (float*)   take((size_t)PPAD * HSZ * 4);
    float*    yb   = (float*)   take((size_t)PPAD * HSZ * 4);
    float*    gin  = (float*)   take((size_t)NN * G3H * 4);
    int2*     prs  = (int2*)    take((size_t)PN * sizeof(int2));
    float*    part = (float*)   take((size_t)2 * 109 * 128 * 4);
    float*    stat = (float*)   take(2 * 4);

    auto cvt = [&](const float* s, _Float16* d, int n) {
        k_cvt<<<(n + 255) / 256, 256, 0, stream>>>(s, d, n);
    };

    // prep
    k_pairs<<<(PN + 255) / 256, 256, 0, stream>>>(prs);
    cvt(Whh, WhhH, G3H * HSZ);
    cvt(W1,  W1H,  HSZ * HSZ);
    cvt(W2,  W2H,  H2 * HSZ);
    cvt(W3,  W3H,  H2 * H2);
    cvt(hid, hA,   PN * HSZ);
    k_gin<<<(NN * G3H + 255) / 256, 256, 0, stream>>>(x, Wih, bih, gin);

    const dim3 gridGRU(PPAD / 32, HSZ / 16);   // 109 x 128
    const dim3 gridM1 (PPAD / 32, HSZ / 32);   // 109 x 64
    const dim3 gridM23(PPAD / 32, H2  / 32);   // 109 x 32

    // GRU x2 (fused GEMM + cell)
    k_gru<<<gridGRU, 32, 0, stream>>>(hA, hid, WhhH, bhh, gin, prs, 0, hS, hB);
    k_gru<<<gridGRU, 32, 0, stream>>>(hB, hS,  WhhH, bhh, gin, prs, 1, hS, hA);

    // layer 1: [PN,2048] @ W1^T -> 2048, relu + full LN
    k_mlp<<<gridM1, 32, 0, stream>>>(hA, W1H, b1, HSZ, HSZ, yb, part);
    k_reduce<<<1, 256, 0, stream>>>(part, 109 * 64, (float)((size_t)PN * HSZ), stat);
    k_ln<<<(PN * HSZ + 255) / 256, 256, 0, stream>>>(yb, ln1w, ln1b, stat, PN * HSZ, hB);

    // layer 2: -> 1024
    k_mlp<<<gridM23, 32, 0, stream>>>(hB, W2H, b2, HSZ, H2, yb, part);
    k_reduce<<<1, 256, 0, stream>>>(part, 109 * 32, (float)((size_t)PN * H2), stat);
    k_ln<<<(PN * H2 + 255) / 256, 256, 0, stream>>>(yb, ln2w, ln2b, stat, PN * H2, hA);

    // layer 3: 1024 -> 1024
    k_mlp<<<gridM23, 32, 0, stream>>>(hA, W3H, b3, H2, H2, yb, part);
    k_reduce<<<1, 256, 0, stream>>>(part, 109 * 32, (float)((size_t)PN * H2), stat);
    k_ln<<<(PN * H2 + 255) / 256, 256, 0, stream>>>(yb, ln3w, ln3b, stat, PN * H2, hB);

    // head + symmetric output
    k_out<<<(NN * NN + 255) / 256, 256, 0, stream>>>(hB, W4, b4, (float*)d_out);
}
